// RelationNetwork_1322849927387
// MI455X (gfx1250) — matrix operand order repaired and checked by reading, not compiled, not run on hardware
//
#include <hip/hip_runtime.h>
#include <hip/hip_bf16.h>

typedef __attribute__((ext_vector_type(16))) __bf16 bf16x16;
typedef __attribute__((ext_vector_type(8)))  __bf16 bf16x8;
typedef __attribute__((ext_vector_type(4)))  __bf16 bf16x4;
typedef __attribute__((ext_vector_type(8)))  float  f32x8;

#define BATCH 4
#define NPTS  256
#define DIM   128
#define H_ENC 512
#define ZDIM  256
#define HDEC  512
#define OUTD  128

// ---- LDS layout for the main kernel ----
// swizzled We2 fragments: [ntile(16)][kstep(16)][lane(32)][elem(16)] bf16
#define LDS_B_BYTES (16 * 16 * 32 * 16 * 2)          // 262144
#define H_ROW_BYTES (H_ENC * 2 + 16)                 // 1040 (pad dodges bank conflicts)
#define LDS_H_OFF   LDS_B_BYTES
#define LDS_TOTAL   (LDS_B_BYTES + 16 * H_ROW_BYTES) // 278784 (<320KB/WG on CDNA5)

// ---------------------------------------------------------------------------
// Kernel 1: A[b,j,h] = X[b,j,:] @ We1[0:128,:], C[b,i,h] = X[b,i,:] @ We1[128:256,:]
// grid: B*N blocks of 128 threads
// ---------------------------------------------------------------------------
__global__ void k_prep_ac(const float* __restrict__ X, const float* __restrict__ We1,
                          float* __restrict__ Aout, float* __restrict__ Cout) {
    __shared__ float xrow[DIM];
    const int bj  = blockIdx.x;
    const int tid = threadIdx.x;
    xrow[tid] = X[bj * DIM + tid];
    __syncthreads();
    #pragma unroll
    for (int cc = 0; cc < 4; ++cc) {
        const int h = tid + cc * 128;
        float sa = 0.f, sc = 0.f;
        for (int d = 0; d < DIM; ++d) {
            const float x = xrow[d];
            sa = fmaf(x, We1[d * H_ENC + h], sa);
            sc = fmaf(x, We1[(DIM + d) * H_ENC + h], sc);
        }
        Aout[bj * H_ENC + h] = sa;
        Cout[bj * H_ENC + h] = sc;
    }
}

// ---------------------------------------------------------------------------
// Kernel 2: swizzle We2 (f32 [512,256]) into bf16 WMMA B-fragments; zero Zp.
// B-frag k order used here (contiguous K, not AMD's table order): lanes 0-15 hold K=0..15 (N=lane),
// lanes 16-31 hold K=16..31 (N=lane-16); 2 consecutive K per VGPR.
// grid: 32 x 256 = 8192 lane-chunks of 16 bf16
// ---------------------------------------------------------------------------
__global__ void k_prep_w2(const float* __restrict__ We2, __bf16* __restrict__ We2f,
                          float* __restrict__ Zp) {
    const int tid = blockIdx.x * 256 + threadIdx.x;   // 0..8191
    if (tid < BATCH * ZDIM) Zp[tid] = 0.f;
    const int l  = tid & 31;
    const int s  = (tid >> 5) & 15;
    const int nt = tid >> 9;                          // 0..15
    const int n  = (nt << 4) + (l & 15);
    const int kbase = (s << 5) + ((l >> 4) << 4);
    __bf16* dst = We2f + tid * 16;
    #pragma unroll
    for (int e = 0; e < 16; ++e)
        dst[e] = (__bf16)We2[(kbase + e) * ZDIM + n];
}

// ---------------------------------------------------------------------------
// Kernel 3 (main): per batch, GEMM [65536 x 512] x [512 x 256] with rows
// h = relu(A[b,j]+C[b,i]+be1) generated on the fly, then relu(.+be2) pooled.
// grid: 1024 WGs x 256 threads (8 waves); WG bgIdx covers 16 pair-tiles.
// Each wave owns 2 N-tiles (32 output columns), all 16 ntiles covered.
// ---------------------------------------------------------------------------
__global__ __launch_bounds__(256, 1)
void rn_pair_gemm_pool(const float* __restrict__ Arow, const float* __restrict__ Crow,
                       const float* __restrict__ be1,  const float* __restrict__ be2,
                       const float4* __restrict__ We2f, float* __restrict__ Zp) {
    extern __shared__ char smem[];
    const int tid   = threadIdx.x;
    const int wave  = tid >> 5;
    const int lane  = tid & 31;
    const int b     = blockIdx.x >> 8;     // 256 WGs per batch
    const int wgIdx = blockIdx.x & 255;

    // ---- stage full swizzled We2 (256KB) into LDS once per WG ----
    {
        float4* dst = (float4*)smem;
        #pragma unroll 4
        for (int c = 0; c < 64; ++c) {
            const int idx = c * 256 + tid;
            dst[idx] = We2f[idx];
        }
    }

    const int m     = lane & 15;           // output row within 16x16 tile
    const int khalf = lane >> 4;           // A-frag K-half selector
    const char* hrow = smem + LDS_H_OFF + m * H_ROW_BYTES;

    const int nt0 = wave * 2;              // this wave's two N-tiles
    const int n0  = nt0 * 16 + m;
    const int n1  = n0 + 16;
    const float bias0 = be2[n0];
    const float bias1 = be2[n1];
    float colsum0 = 0.f, colsum1 = 0.f;

    __syncthreads();

    for (int it = 0; it < 16; ++it) {
        const int t  = wgIdx * 16 + it;    // pair-tile within batch [0,4096)
        const int i  = t >> 4;             // fixed i for the 16 rows
        const int j0 = (t & 15) << 4;      // rows m -> j = j0 + m

        // ---- build h tile (16 x 512) in bf16 into LDS, 256 threads ----
        {
            const float* Cp = Crow + ((b << 8) + i) * H_ENC;
            #pragma unroll
            for (int cc = 0; cc < 8; ++cc) {
                const int chunk = cc * 256 + tid;   // 0..2047 float4-chunks
                const int row   = chunk >> 7;
                const int c4    = chunk & 127;
                const float4 av = *(const float4*)(Arow + ((((b << 8) + j0 + row) << 9) + (c4 << 2)));
                const float4 cv = *(const float4*)(Cp + (c4 << 2));
                const float4 bv = *(const float4*)(be1 + (c4 << 2));
                bf16x4 o;
                o.x = (__bf16)fmaxf(av.x + cv.x + bv.x, 0.f);
                o.y = (__bf16)fmaxf(av.y + cv.y + bv.y, 0.f);
                o.z = (__bf16)fmaxf(av.z + cv.z + bv.z, 0.f);
                o.w = (__bf16)fmaxf(av.w + cv.w + bv.w, 0.f);
                *(bf16x4*)(smem + LDS_H_OFF + row * H_ROW_BYTES + (c4 << 3)) = o;
            }
        }
        __syncthreads();

        // ---- 16 K-steps of v_wmma_f32_16x16x32_bf16, 2 N-tiles per wave ----
        f32x8 acc0 = {};
        f32x8 acc1 = {};
        #pragma unroll
        for (int s = 0; s < 16; ++s) {
            // A-frag (16x32 bf16): lane half selects K 0-15 vs 16-31
            const int k0 = (s << 5) + (khalf << 4);
            union { bf16x16 v; struct { bf16x8 lo, hi; } h; } a;
            a.h.lo = *(const bf16x8*)(hrow + k0 * 2);
            a.h.hi = *(const bf16x8*)(hrow + k0 * 2 + 16);
            // B-frags: pre-swizzled, 32 contiguous bytes per lane
            const bf16x16 bf0 = *(const bf16x16*)(smem + (((nt0    ) * 16 + s) * 32 + lane) * 32);
            const bf16x16 bf1 = *(const bf16x16*)(smem + (((nt0 + 1) * 16 + s) * 32 + lane) * 32);
            acc0 = __builtin_amdgcn_wmma_f32_16x16x32_bf16(false, a.v, false, bf0,
                                                           (short)0, acc0, false, false);
            acc1 = __builtin_amdgcn_wmma_f32_16x16x32_bf16(false, a.v, false, bf1,
                                                           (short)0, acc1, false, false);
        }

        // ---- bias + ReLU + pool the 16 pair-rows of this tile ----
        #pragma unroll
        for (int r = 0; r < 8; ++r) {
            colsum0 += fmaxf(acc0[r] + bias0, 0.f);
            colsum1 += fmaxf(acc1[r] + bias1, 0.f);
        }
        __syncthreads();   // protect h tile before next overwrite
    }

    // rows M=0..7 live in lanes 0-15's regs, M=8..15 in lanes 16-31: fold halves
    colsum0 += __shfl_xor(colsum0, 16, 32);
    colsum1 += __shfl_xor(colsum1, 16, 32);
    if (lane < 16) {
        atomicAdd(&Zp[(b << 8) + n0], colsum0);
        atomicAdd(&Zp[(b << 8) + n1], colsum1);
    }
}

// ---------------------------------------------------------------------------
// Kernel 4: decoder. hd = relu(Zp @ Wd1 + bd1); out = hd @ Wd2 + bd2
// single WG of 512 threads
// ---------------------------------------------------------------------------
__global__ void k_dec(const float* __restrict__ Zp,  const float* __restrict__ Wd1,
                      const float* __restrict__ bd1, const float* __restrict__ Wd2,
                      const float* __restrict__ bd2, float* __restrict__ out) {
    __shared__ float zp_s[BATCH * ZDIM];
    __shared__ float hd_s[BATCH * HDEC];
    const int tid = threadIdx.x;
    for (int q = tid; q < BATCH * ZDIM; q += 512) zp_s[q] = Zp[q];
    __syncthreads();
    #pragma unroll
    for (int bb = 0; bb < BATCH; ++bb) {
        float s = bd1[tid];
        for (int k = 0; k < ZDIM; ++k)
            s = fmaf(zp_s[bb * ZDIM + k], Wd1[k * HDEC + tid], s);
        hd_s[bb * HDEC + tid] = fmaxf(s, 0.f);
    }
    __syncthreads();
    {
        const int bb = tid >> 7, o = tid & 127;   // 4*128 = 512 outputs
        float s = bd2[o];
        for (int k = 0; k < HDEC; ++k)
            s = fmaf(hd_s[bb * HDEC + k], Wd2[k * OUTD + o], s);
        out[bb * OUTD + o] = s;
    }
}

// ---------------------------------------------------------------------------
extern "C" void kernel_launch(void* const* d_in, const int* in_sizes, int n_in,
                              void* d_out, int out_size, void* d_ws, size_t ws_size,
                              hipStream_t stream) {
    const float* X   = (const float*)d_in[0];
    const float* We1 = (const float*)d_in[1];
    const float* be1 = (const float*)d_in[2];
    const float* We2 = (const float*)d_in[3];
    const float* be2 = (const float*)d_in[4];
    const float* Wd1 = (const float*)d_in[5];
    const float* bd1 = (const float*)d_in[6];
    const float* Wd2 = (const float*)d_in[7];
    const float* bd2 = (const float*)d_in[8];
    float* out = (float*)d_out;

    // workspace layout: A (2MB) | C (2MB) | We2f bf16 frags (256KB) | Zp (4KB)
    char*   ws   = (char*)d_ws;
    float*  Aout = (float*)ws;
    float*  Cout = (float*)(ws + (2u << 20));
    __bf16* We2f = (__bf16*)(ws + (4u << 20));
    float*  Zp   = (float*)(ws + (4u << 20) + (256u << 10));

    hipFuncSetAttribute((const void*)rn_pair_gemm_pool,
                        hipFuncAttributeMaxDynamicSharedMemorySize, LDS_TOTAL);

    k_prep_ac<<<BATCH * NPTS, 128, 0, stream>>>(X, We1, Aout, Cout);
    k_prep_w2<<<32, 256, 0, stream>>>(We2, We2f, Zp);
    rn_pair_gemm_pool<<<1024, 256, LDS_TOTAL, stream>>>(Aout, Cout, be1, be2,
                                                        (const float4*)We2f, Zp);
    k_dec<<<1, 512, 0, stream>>>(Zp, Wd1, bd1, Wd2, bd2, out);
}
